// PolyNetFP4Sim_76355928588536
// MI455X (gfx1250) — matrix operand order repaired and checked by reading, not compiled, not run on hardware
//
#include <hip/hip_runtime.h>

// ---------------------------------------------------------------------------
// Fused 4-layer MLP with simulated FP4 weight quantization (MI455X / gfx1250)
//   h1 = silu(x @ q(w1).T + b1)        (B,1)->(B,64)   VALU outer product
//   h2 = silu(h1 @ q(w2).T + b2)       (B,64)->(B,64)  WMMA f16 16x16x32
//   h3 = silu(h2 @ q(w3).T + b3)       (B,64)->(B,32)  WMMA f16 16x16x32
//   y  = h3 @ q(w4).T + b4             (B,32)->(B,1)   VALU dot + shuffle
// Computed transposed: features on M (weights = A operand, per-wave constant),
// samples on N (activations = B operand).  Inter-layer handoff needs only a
// lane<->lane^16 exchange (same sample lives in lanes l and l^16).
// Round 3: SiLU evaluated in explicit batched stages (8-16 independent
// exp/rcp per stage) so TRANS latency is pipelined with no v_nop hazard
// filler; no hard waves-per-EU cap (round 2's cap serialized the chains).
// ---------------------------------------------------------------------------

typedef __attribute__((ext_vector_type(16))) _Float16 v16h;
typedef __attribute__((ext_vector_type(8)))  _Float16 v8h;
typedef __attribute__((ext_vector_type(8)))  float    v8f;

union V16HU { v16h v; unsigned u[8]; };
union H2U   { _Float16 h[2]; unsigned u; };

// 1-2-1 FP4 (exp bias 1): mant,exp = frexp(|w|), qe = clip(e+1,0,3),
// qm = (mant >= 0.75), decoded = sign * ldexp(qm ? 0.75 : 0.5, qe-1)
__device__ __forceinline__ float qfp4(float w) {
    if (w == 0.0f) return 0.0f;
    int e;
    float m  = frexpf(fabsf(w), &e);
    int  qe  = e + 1;
    qe = qe < 0 ? 0 : (qe > 3 ? 3 : qe);
    float dec = ldexpf(m >= 0.75f ? 0.75f : 0.5f, qe - 1);
    return w < 0.0f ? -dec : dec;
}

// ---- weight prep: quantize all weights to f16 into workspace --------------
// ws layout (halves): w1q[64] | w2q[4096] | w3q[2048] | w4q[32]   (12480 B)
__global__ void quant_prep_kernel(const float* __restrict__ w1,
                                  const float* __restrict__ w2,
                                  const float* __restrict__ w3,
                                  const float* __restrict__ w4,
                                  _Float16* __restrict__ ws) {
    for (int t = blockIdx.x * blockDim.x + threadIdx.x; t < 6240;
         t += blockDim.x * gridDim.x) {
        float v;
        if      (t < 64)   v = w1[t];
        else if (t < 4160) v = w2[t - 64];
        else if (t < 6208) v = w3[t - 4160];
        else               v = w4[t - 6208];
        ws[t] = (_Float16)qfp4(v);
    }
}

__global__ void __launch_bounds__(256) mlp_fused_kernel(
    const float* __restrict__ x,
    const float* __restrict__ b1, const float* __restrict__ b2,
    const float* __restrict__ b3, const float* __restrict__ b4,
    const _Float16* __restrict__ wsq,
    float* __restrict__ out, int nTiles, int nWaves)
{
    const _Float16* w1q = wsq;
    const _Float16* w2q = wsq + 64;
    const _Float16* w3q = wsq + 4160;
    const _Float16* w4q = wsq + 6208;

    const int lane = threadIdx.x & 31;
    const int row  = lane & 15;      // sample slot / A-row slot
    const int h    = lane >> 4;      // wave half

    // ---- per-wave constant fragments (built once, reused over many tiles) --
    // Layer-1 params packed f16 {w, b}: slot j -> k = (j/16)*32 + 16h + j%16.
    // w1 is FP4-quantized => exact in f16.  fmaf(f32,(float)f16,(float)f16)
    // lowers to v_fma_mix_f32.
    H2U wb1[32];
#pragma unroll
    for (int j = 0; j < 32; ++j) {
        int k = ((j >> 4) << 5) + h * 16 + (j & 15);
        wb1[j].h[0] = w1q[k];
        wb1[j].h[1] = (_Float16)b1[k];
    }

    // Layer-2 weights as A fragments: a2w[ma][kb] = 16x32 tile of q(w2)
    // A layout: lane row = ma*16+row; elem t -> k = kb*32 + 16h + t
    //           (two consecutive 16B runs)
    v16h a2w[4][2];
#pragma unroll
    for (int ma = 0; ma < 4; ++ma)
#pragma unroll
        for (int kb = 0; kb < 2; ++kb) {
            const v8h* p = (const v8h*)(w2q + (ma * 16 + row) * 64 + kb * 32 + h * 16);
            v8h lo = p[0];
            v8h hi = p[1];   // +8 halves
            a2w[ma][kb] = __builtin_shufflevector(lo, hi,
                0,1,2,3,4,5,6,7,8,9,10,11,12,13,14,15);
        }

    // Layer-3 weights as A fragments (32x64 -> 2 M-tiles x 2 K-tiles)
    v16h a3w[2][2];
#pragma unroll
    for (int ma = 0; ma < 2; ++ma)
#pragma unroll
        for (int kb = 0; kb < 2; ++kb) {
            const v8h* p = (const v8h*)(w3q + (ma * 16 + row) * 64 + kb * 32 + h * 16);
            v8h lo = p[0];
            v8h hi = p[1];
            a3w[ma][kb] = __builtin_shufflevector(lo, hi,
                0,1,2,3,4,5,6,7,8,9,10,11,12,13,14,15);
        }

    // Bias accumulator-init fragments (f32: fed straight into WMMA C operand)
    // C[m][n] = bias[feature], feature = ma*16 + 8h + v
    v8f c2[4];
#pragma unroll
    for (int ma = 0; ma < 4; ++ma)
#pragma unroll
        for (int v = 0; v < 8; ++v) c2[ma][v] = b2[ma * 16 + 8 * h + v];
    v8f c3[2];
#pragma unroll
    for (int ma = 0; ma < 2; ++ma)
#pragma unroll
        for (int v = 0; v < 8; ++v) c3[ma][v] = b3[ma * 16 + 8 * h + v];

    // Layer-4 weights matching the D3 layout (packed f16 pairs, fma_mix):
    // slot j = ma*8+v -> feature = ma*16 + 8h + (j&7)
    H2U w4p[8];
#pragma unroll
    for (int jj = 0; jj < 8; ++jj) {
        int j0 = 2 * jj, j1 = 2 * jj + 1;
        w4p[jj].h[0] = w4q[((j0 >> 3) << 4) + 8 * h + (j0 & 7)];
        w4p[jj].h[1] = w4q[((j1 >> 3) << 4) + 8 * h + (j1 & 7)];
    }
    const float b4s = b4[0];

    const int wave0 = blockIdx.x * (blockDim.x >> 5) + (threadIdx.x >> 5);

    // ---- grid-stride loop over 16-sample tiles ----------------------------
    for (int T = wave0; T < nTiles; T += nWaves) {
        const int base = T << 4;
        const float xs = x[base + row];   // lanes l and l^16 share a sample

        // ---- Layer 1: B fragments of h1^T, SiLU in batched stages ---------
        // B layout: lane col = row, elem t -> k = kb*32 + 16h + t
        v16h bf[2];
#pragma unroll
        for (int kb = 0; kb < 2; ++kb) {
            float pre[16], r[16];
#pragma unroll
            for (int t = 0; t < 16; ++t) {
                const H2U q = wb1[kb * 16 + t];
                pre[t] = fmaf(xs, (float)q.h[0], (float)q.h[1]);
            }
#pragma unroll
            for (int t = 0; t < 16; ++t) r[t] = __expf(-pre[t]);   // 16 indep exps
#pragma unroll
            for (int t = 0; t < 16; ++t) r[t] = 1.0f + r[t];
#pragma unroll
            for (int t = 0; t < 16; ++t) r[t] = __builtin_amdgcn_rcpf(r[t]);
            v16h f;
#pragma unroll
            for (int t = 0; t < 16; ++t) f[t] = (_Float16)(pre[t] * r[t]);
            bf[kb] = f;
        }

        // ---- Layer 2: D2[ma] = q(w2)[ma-tile] x h1^T + b2  (4 M-tiles) ----
        unsigned pk[4][4];   // silu(h2) as packed f16 pairs, D layout
#pragma unroll
        for (int ma = 0; ma < 4; ++ma) {
            v8f acc = c2[ma];
            acc = __builtin_amdgcn_wmma_f32_16x16x32_f16(
                false, a2w[ma][0], false, bf[0], (short)0, acc, false, false);
            acc = __builtin_amdgcn_wmma_f32_16x16x32_f16(
                false, a2w[ma][1], false, bf[1], (short)0, acc, false, false);
            float r[8];
#pragma unroll
            for (int v = 0; v < 8; ++v) r[v] = __expf(-acc[v]);    // 8 indep exps
#pragma unroll
            for (int v = 0; v < 8; ++v) r[v] = 1.0f + r[v];
#pragma unroll
            for (int v = 0; v < 8; ++v) r[v] = __builtin_amdgcn_rcpf(r[v]);
#pragma unroll
            for (int j = 0; j < 4; ++j) {
                H2U u;                                  // v_fma_mix{lo,hi}_f16
                u.h[0] = (_Float16)(acc[2 * j] * r[2 * j]);
                u.h[1] = (_Float16)(acc[2 * j + 1] * r[2 * j + 1]);
                pk[ma][j] = u.u;
            }
        }

        // Redistribute D2 -> layer-3 B fragments. Lanes l and l^16 hold the
        // same sample with complementary feature halves: one shfl_xor(16).
        // Dest lane (half h) needs M-tile (2kb+h): t<8 from the half-0 lane,
        // t>=8 from the half-1 lane.  keep = pk[2kb+h], send = pk[2kb+1-h].
        v16h b3f[2];
#pragma unroll
        for (int kb = 0; kb < 2; ++kb) {
            V16HU u;
#pragma unroll
            for (int j = 0; j < 4; ++j) {
                unsigned keep = h ? pk[2 * kb + 1][j] : pk[2 * kb][j];
                unsigned sent = h ? pk[2 * kb][j]     : pk[2 * kb + 1][j];
                unsigned recv = (unsigned)__shfl_xor((int)sent, 16, 32);
                u.u[j]     = h ? recv : keep;
                u.u[4 + j] = h ? keep : recv;
            }
            b3f[kb] = u.v;
        }

        // ---- Layer 3 (+ fused layer-4 partial dot, two accumulators) ------
        float p0 = 0.0f, p1 = 0.0f;
#pragma unroll
        for (int ma = 0; ma < 2; ++ma) {
            v8f acc = c3[ma];
            acc = __builtin_amdgcn_wmma_f32_16x16x32_f16(
                false, a3w[ma][0], false, b3f[0], (short)0, acc, false, false);
            acc = __builtin_amdgcn_wmma_f32_16x16x32_f16(
                false, a3w[ma][1], false, b3f[1], (short)0, acc, false, false);
            float r[8];
#pragma unroll
            for (int v = 0; v < 8; ++v) r[v] = __expf(-acc[v]);
#pragma unroll
            for (int v = 0; v < 8; ++v) r[v] = 1.0f + r[v];
#pragma unroll
            for (int v = 0; v < 8; ++v) r[v] = __builtin_amdgcn_rcpf(r[v]);
#pragma unroll
            for (int v = 0; v < 8; ++v) {
                const float hv = acc[v] * r[v];       // silu(h3)
                const float wv = (float)w4p[ma * 4 + (v >> 1)].h[v & 1];
                if (v & 1) p1 = fmaf(hv, wv, p1);
                else       p0 = fmaf(hv, wv, p0);
            }
        }
        float p = p0 + p1;

        // ---- Layer 4: reduce the two feature halves, store f32 ------------
        float tot = p + __shfl_xor(p, 16, 32);
        if (h == 0) out[base + row] = tot + b4s;
    }
}

extern "C" void kernel_launch(void* const* d_in, const int* in_sizes, int n_in,
                              void* d_out, int out_size, void* d_ws, size_t ws_size,
                              hipStream_t stream) {
    const float* x  = (const float*)d_in[0];
    const float* w1 = (const float*)d_in[1];
    const float* b1 = (const float*)d_in[2];
    const float* w2 = (const float*)d_in[3];
    const float* b2 = (const float*)d_in[4];
    const float* w3 = (const float*)d_in[5];
    const float* b3 = (const float*)d_in[6];
    const float* w4 = (const float*)d_in[7];
    const float* b4 = (const float*)d_in[8];
    float*       out = (float*)d_out;
    _Float16*    wsq = (_Float16*)d_ws;

    // 1) quantize weights (tiny; stays resident in L2)
    quant_prep_kernel<<<4, 256, 0, stream>>>(w1, w2, w3, w4, wsq);

    // 2) fused MLP: 1024 blocks x 8 waves, grid-stride over 16-sample tiles
    const int B      = in_sizes[0];       // 1048576, multiple of 16
    const int nTiles = B >> 4;
    const int blocks = 1024;
    const int nWaves = blocks * (256 / 32);
    mlp_fused_kernel<<<blocks, 256, 0, stream>>>(x, b1, b2, b3, b4, wsq, out,
                                                 nTiles, nWaves);
}
